// MultiHeadHistory_TargetAttention_10196252360984
// MI455X (gfx1250) — compile-verified, matrix operand order checked by reading, not run on hardware
//
#include <hip/hip_runtime.h>
#include <hip/hip_bf16.h>
#include <stdint.h>

// Problem constants
#define Bq 8
#define Sq 1024
#define Eq 1024
#define Hq 16
#define Dq 64

typedef __attribute__((ext_vector_type(16))) __bf16 v16bf;
typedef __attribute__((ext_vector_type(8)))  float  v8f;
typedef __attribute__((ext_vector_type(4)))  unsigned int v4u;
typedef __attribute__((ext_vector_type(8)))  int v8i;
typedef __attribute__((ext_vector_type(4)))  int v4i;

union Frag {
    v16bf v;
    unsigned short u[16];
    uint4 q[2];
};

__device__ __forceinline__ unsigned short f2bf(float f) {
    union { float f; unsigned u; } c; c.f = f;
    unsigned u = c.u;
    u += 0x7fffu + ((u >> 16) & 1u);   // round-to-nearest-even
    return (unsigned short)(u >> 16);
}
__device__ __forceinline__ float bf2f(unsigned short h) {
    union { unsigned u; float f; } c; c.u = ((unsigned)h) << 16;
    return c.f;
}

__device__ __forceinline__ uint32_t lds_offset(const void* p) {
    // addrspace(3) flat address: low 32 bits are the LDS byte offset
    return (uint32_t)(uintptr_t)p;
}

// ---------------------------------------------------------------------------
// Tensor Data Mover: async 2D bf16 tile load Global -> LDS.
// D# per CDNA5 ISA ch.8: group0 = {count/flags, lds_addr, global_addr, type},
// group1 = {data_size/padding, tensor dims, tile dims, strides}.
// pad_interval=3 (16 DWORDs = 64B = one 32-elem bf16 row), pad_amount=3
// (4 DWORDs = 16B) -> hardware produces the 80B (40-half) padded LDS stride.
// ---------------------------------------------------------------------------
__device__ __forceinline__ void tdm_load_2d_bf16(
    const void* gptr, uint32_t ldsAddr, int tileW, int tileH, int strideElems)
{
    uint64_t ga = (uint64_t)(uintptr_t)gptr;
    v4u g0;
    g0[0] = 1u;                                             // count=1, no gather
    g0[1] = ldsAddr;                                        // LDS byte address
    g0[2] = (uint32_t)ga;                                   // global_addr[31:0]
    g0[3] = (uint32_t)((ga >> 32) & 0x1FFFFFFu) | (2u << 30); // [56:32] | type=2

    const uint32_t td0 = 1u << 20, td1 = 1u << 20;          // generous tensor dims
    v8i g1;
    g1[0] = (int)((1u << 16) |        // data_size = 1 -> 2 bytes
                  (1u << 20) |        // pad_enable
                  (3u << 22) |        // pad_interval: every 16 DWORDs (64B)
                  (3u << 25));        // pad_amount: 4 DWORDs (16B)
    g1[1] = (int)((td0 & 0xFFFFu) << 16);                   // tensor_dim0[15:0]
    g1[2] = (int)((td0 >> 16) | ((td1 & 0xFFFFu) << 16));   // td0 hi | td1 lo
    g1[3] = (int)((td1 >> 16) | ((uint32_t)tileW << 16));   // td1 hi | tile_dim0
    g1[4] = (int)(uint32_t)tileH;                           // tile_dim1 (dim2=0)
    g1[5] = strideElems;                                    // tensor_dim0_stride lo32
    g1[6] = 0;                                              // stride0 hi | stride1 lo
    g1[7] = 0;                                              // stride1 hi

    v4i z4 = {0, 0, 0, 0};
#if __clang_major__ >= 23
    v8i z8 = {0, 0, 0, 0, 0, 0, 0, 0};
    __builtin_amdgcn_tensor_load_to_lds(g0, g1, z4, z4, z8, 0);
#else
    __builtin_amdgcn_tensor_load_to_lds(g0, g1, z4, z4, 0);
#endif
}

// ---------------------------------------------------------------------------
// fp32 -> bf16 elementwise convert (bandwidth bound, float4-vectorized)
// ---------------------------------------------------------------------------
__global__ __launch_bounds__(256) void cvt_kernel(
    const float* __restrict__ in, unsigned short* __restrict__ out, int n4)
{
    int i = blockIdx.x * blockDim.x + threadIdx.x;
    if (i < n4) {
        float4 f = ((const float4*)in)[i];
        ushort4 h; h.x = f2bf(f.x); h.y = f2bf(f.y); h.z = f2bf(f.z); h.w = f2bf(f.w);
        ((ushort4*)out)[i] = h;
    }
}

// ---------------------------------------------------------------------------
// Unified bf16 GEMM: Y = A @ Wb^T + bias  (M=8192, N=1024, K=1024)
// A bf16 [M][K], Wb bf16 [N][K].  Block tile 64(M) x 128(N), 8 waves,
// 2x2 WMMA accumulators per wave.  A/B tiles DMA'd by the Tensor Data Mover
// (per-wave slices), double-buffered on TENSORcnt.
// Output: Yh = bf16 head layout [b][h][s][d], Yt = bf16 transposed
// [b][h][d][s], Yf = fp32 row-major (whichever are non-null).
// ---------------------------------------------------------------------------
__global__ __launch_bounds__(256) void gemm_bf16(
    const unsigned short* __restrict__ A, const unsigned short* __restrict__ Wb,
    const float* __restrict__ bias,
    unsigned short* __restrict__ Yh, unsigned short* __restrict__ Yt,
    float* __restrict__ Yf)
{
    __shared__ __align__(16) unsigned short As[2][64 * 40];   // 80B padded rows
    __shared__ __align__(16) unsigned short Bs[2][128 * 40];

    const int tid   = threadIdx.x;
    const int nBase = blockIdx.x * 128;
    const int mBase = blockIdx.y * 64;
    const int lane  = tid & 31, wave = tid >> 5;
    const int hl    = lane & 15, hi = lane >> 4;
    const int mw    = wave & 1,  nw = wave >> 1;

    const int nK = Eq / 32;   // 32 K-steps

    // Prologue: DMA tiles for k-steps 0 and 1 (each wave moves its own slice)
#pragma unroll
    for (int pre = 0; pre < 2; ++pre) {
        int k0 = pre * 32;
        tdm_load_2d_bf16(A + (size_t)(mBase + wave * 8) * Eq + k0,
                         lds_offset(&As[pre][wave * 8 * 40]), 32, 8, Eq);
        tdm_load_2d_bf16(Wb + (size_t)(nBase + wave * 16) * Eq + k0,
                         lds_offset(&Bs[pre][wave * 16 * 40]), 32, 16, Eq);
    }

    v8f acc[2][2] = {};

    for (int k = 0; k < nK; ++k) {
        const int p = k & 1;
        // Wait for this iteration's 2 DMAs (in-order; next iter's may remain)
        if (k + 1 < nK) __builtin_amdgcn_s_wait_tensorcnt(2);
        else            __builtin_amdgcn_s_wait_tensorcnt(0);
        __syncthreads();

        Frag a[2], b[2];
#pragma unroll
        for (int mi = 0; mi < 2; ++mi) {
            const unsigned short* pp = &As[p][(mw * 32 + mi * 16 + hl) * 40 + hi * 8];
            a[mi].q[0] = *(const uint4*)pp;          // K = koff .. koff+7
            a[mi].q[1] = *(const uint4*)(pp + 16);   // K = koff+16 .. koff+23
        }
#pragma unroll
        for (int ni = 0; ni < 2; ++ni) {
            const unsigned short* pp = &Bs[p][(nw * 32 + ni * 16 + hl) * 40 + hi * 8];
            b[ni].q[0] = *(const uint4*)pp;
            b[ni].q[1] = *(const uint4*)(pp + 16);
        }
#pragma unroll
        for (int mi = 0; mi < 2; ++mi)
#pragma unroll
            for (int ni = 0; ni < 2; ++ni)
                acc[mi][ni] = __builtin_amdgcn_wmma_f32_16x16x32_bf16(
                    false, a[mi].v, false, b[ni].v, (short)0, acc[mi][ni], false, false);

        __syncthreads();   // all waves done reading buffer p before refilling it

        if (k + 2 < nK) {
            int k0 = (k + 2) * 32;
            tdm_load_2d_bf16(A + (size_t)(mBase + wave * 8) * Eq + k0,
                             lds_offset(&As[p][wave * 8 * 40]), 32, 8, Eq);
            tdm_load_2d_bf16(Wb + (size_t)(nBase + wave * 16) * Eq + k0,
                             lds_offset(&Bs[p][wave * 16 * 40]), 32, 16, Eq);
        }
    }

    // Epilogue
#pragma unroll
    for (int mi = 0; mi < 2; ++mi) {
#pragma unroll
        for (int ni = 0; ni < 2; ++ni) {
            int gn = nBase + nw * 32 + ni * 16 + hl;
            float bv = bias[gn];
            int h = gn >> 6, dc = gn & 63;
#pragma unroll
            for (int i = 0; i < 8; ++i) {
                int gm = mBase + mw * 32 + mi * 16 + hi * 8 + i;
                float val = acc[mi][ni][i] + bv;
                if (Yf) {
                    Yf[(size_t)gm * Eq + gn] = val;
                } else {
                    int bb = gm >> 10, ss = gm & 1023;
                    unsigned short o = f2bf(val);
                    if (Yh) Yh[(((size_t)(bb * Hq + h) * Sq + ss) << 6) + dc] = o;
                    if (Yt) Yt[((((size_t)(bb * Hq + h)) << 6) + dc) * Sq + ss] = o;
                }
            }
        }
    }
}

// ---------------------------------------------------------------------------
// Attention: one (b,h, 16-query block) per workgroup (4 waves / 128 threads).
// scores = Q K^T * 1/8 -> softmax -> write fp32 attn -> context = P V.
// ---------------------------------------------------------------------------
__global__ __launch_bounds__(128) void attn_kernel(
    const unsigned short* __restrict__ Qp, const unsigned short* __restrict__ Kp,
    const unsigned short* __restrict__ Vt, float* __restrict__ attn,
    unsigned short* __restrict__ Ctx)
{
    __shared__ __align__(16) unsigned short sc[16 * 1040];   // padded bf16 rows
    __shared__ float red[16][8];
    __shared__ float rmax[16];
    __shared__ float rsum[16];

    const int tid  = threadIdx.x;
    const int lane = tid & 31, wave = tid >> 5;
    const int hl   = lane & 15, hi = lane >> 4;
    const int bh   = blockIdx.x >> 6;
    const int q0   = (blockIdx.x & 63) << 4;

    const unsigned short* Qb = Qp + (size_t)bh * Sq * Dq;
    const unsigned short* Kb = Kp + (size_t)bh * Sq * Dq;
    const unsigned short* Vb = Vt + (size_t)bh * Dq * Sq;

    Frag qa[2];
#pragma unroll
    for (int kk = 0; kk < 2; ++kk) {
        const unsigned short* p = Qb + (q0 + hl) * Dq + kk * 32 + hi * 8;
        qa[kk].q[0] = *(const uint4*)p;
        qa[kk].q[1] = *(const uint4*)(p + 16);
    }

    const float scale = 0.125f;   // 1/sqrt(64)
    for (int t = 0; t < 16; ++t) {
        int s0 = (wave + t * 4) << 4;
        v8f a = {};
#pragma unroll
        for (int kk = 0; kk < 2; ++kk) {
            Frag kf;
            const unsigned short* p = Kb + (s0 + hl) * Dq + kk * 32 + hi * 8;
            kf.q[0] = *(const uint4*)p;
            kf.q[1] = *(const uint4*)(p + 16);
            a = __builtin_amdgcn_wmma_f32_16x16x32_bf16(
                false, qa[kk].v, false, kf.v, (short)0, a, false, false);
        }
#pragma unroll
        for (int i = 0; i < 8; ++i)
            sc[(hi * 8 + i) * 1040 + s0 + hl] = f2bf(a[i] * scale);
    }
    __syncthreads();

    // Row-wise softmax: 8 threads per row, 128 cols each
    const int row = tid >> 3, seg = tid & 7;
    unsigned short* rp = sc + row * 1040 + seg * 128;
    float mx = -3.0e38f;
    for (int j = 0; j < 128; ++j) mx = fmaxf(mx, bf2f(rp[j]));
    red[row][seg] = mx;
    __syncthreads();
    if (seg == 0) {
        float m = red[row][0];
#pragma unroll
        for (int k = 1; k < 8; ++k) m = fmaxf(m, red[row][k]);
        rmax[row] = m;
    }
    __syncthreads();
    const float m = rmax[row];
    float sum = 0.f;
    for (int j = 0; j < 128; ++j) {
        float e = __expf(bf2f(rp[j]) - m);
        rp[j] = f2bf(e);
        sum += e;
    }
    red[row][seg] = sum;
    __syncthreads();
    if (seg == 0) {
        float s = 0.f;
#pragma unroll
        for (int k = 0; k < 8; ++k) s += red[row][k];
        rsum[row] = s;
    }
    __syncthreads();
    const float inv = 1.0f / rsum[row];
    for (int j = 0; j < 128; ++j) rp[j] = f2bf(bf2f(rp[j]) * inv);
    __syncthreads();

    // Coalesced fp32 attn_weights writeback (float4)
    float* aout = attn + ((size_t)bh * Sq + q0) * Sq;
    for (int j = 0; j < 32; ++j) {
        int lin = tid + j * 128;
        int r = lin >> 8, c4 = lin & 255;
        ushort4 hq = *(const ushort4*)(sc + r * 1040 + c4 * 4);
        float4 o; o.x = bf2f(hq.x); o.y = bf2f(hq.y); o.z = bf2f(hq.z); o.w = bf2f(hq.w);
        *(float4*)(aout + (size_t)r * Sq + c4 * 4) = o;
    }

    // context = P @ V : per-wave 16x16 output tile (d0 = wave*16)
    const int d0 = wave << 4;
    v8f c = {};
    for (int st = 0; st < 32; ++st) {
        int sb = st << 5;
        Frag pa, vf;
        const unsigned short* pp = sc + hl * 1040 + sb + hi * 8;
        pa.q[0] = *(const uint4*)pp;
        pa.q[1] = *(const uint4*)(pp + 16);
        const unsigned short* vp = Vb + (d0 + hl) * Sq + sb + hi * 8;  // Vt[d][s]
        vf.q[0] = *(const uint4*)vp;
        vf.q[1] = *(const uint4*)(vp + 16);
        c = __builtin_amdgcn_wmma_f32_16x16x32_bf16(
            false, pa.v, false, vf.v, (short)0, c, false, false);
    }
    const int bb = bh >> 4, h = bh & 15;
#pragma unroll
    for (int i = 0; i < 8; ++i) {
        int q = q0 + hi * 8 + i;
        Ctx[(((size_t)(bb * Sq + q)) << 10) + (h << 6) + d0 + hl] = f2bf(c[i]);
    }
}

// ---------------------------------------------------------------------------
extern "C" void kernel_launch(void* const* d_in, const int* in_sizes, int n_in,
                              void* d_out, int out_size, void* d_ws, size_t ws_size,
                              hipStream_t stream) {
    const float* query = (const float*)d_in[0];
    const float* key_  = (const float*)d_in[1];
    const float* value = (const float*)d_in[2];
    const float* q_w   = (const float*)d_in[3];
    const float* q_b   = (const float*)d_in[4];
    const float* k_w   = (const float*)d_in[5];
    const float* k_b   = (const float*)d_in[6];
    const float* v_w   = (const float*)d_in[7];
    const float* v_b   = (const float*)d_in[8];
    const float* out_w = (const float*)d_in[9];
    const float* out_b = (const float*)d_in[10];

    uint8_t* ws = (uint8_t*)d_ws;
    unsigned short* Qp  = (unsigned short*)(ws);                        // 16 MB
    unsigned short* Kp  = (unsigned short*)(ws + ((size_t)16 << 20));   // 16 MB
    unsigned short* Vt  = (unsigned short*)(ws + ((size_t)32 << 20));   // 16 MB
    unsigned short* Ctx = (unsigned short*)(ws + ((size_t)48 << 20));   // 16 MB
    unsigned short* Xq  = (unsigned short*)(ws + ((size_t)64 << 20));   // 16 MB
    unsigned short* Xk  = (unsigned short*)(ws + ((size_t)80 << 20));   // 16 MB
    unsigned short* Xv  = (unsigned short*)(ws + ((size_t)96 << 20));   // 16 MB
    unsigned short* Wqb = (unsigned short*)(ws + ((size_t)112 << 20));  // 2 MB
    unsigned short* Wkb = (unsigned short*)(ws + ((size_t)114 << 20));  // 2 MB
    unsigned short* Wvb = (unsigned short*)(ws + ((size_t)116 << 20));  // 2 MB
    unsigned short* Wob = (unsigned short*)(ws + ((size_t)118 << 20));  // 2 MB

    float* attn = (float*)d_out;                                        // (B,H,S,S)
    float* outp = attn + (size_t)Bq * Hq * Sq * Sq;                     // (B,S,E)

    const int n4x = (Bq * Sq * Eq) / 4;   // 2M float4 per activation tensor
    const int n4w = (Eq * Eq) / 4;        // 256K float4 per weight matrix
    cvt_kernel<<<n4x / 256, 256, 0, stream>>>(query, Xq, n4x);
    cvt_kernel<<<n4x / 256, 256, 0, stream>>>(key_,  Xk, n4x);
    cvt_kernel<<<n4x / 256, 256, 0, stream>>>(value, Xv, n4x);
    cvt_kernel<<<n4w / 256, 256, 0, stream>>>(q_w,   Wqb, n4w);
    cvt_kernel<<<n4w / 256, 256, 0, stream>>>(k_w,   Wkb, n4w);
    cvt_kernel<<<n4w / 256, 256, 0, stream>>>(v_w,   Wvb, n4w);
    cvt_kernel<<<n4w / 256, 256, 0, stream>>>(out_w, Wob, n4w);

    dim3 gg(Eq / 128, (Bq * Sq) / 64);
    gemm_bf16<<<gg, 256, 0, stream>>>(Xq, Wqb, q_b, Qp, nullptr, nullptr);
    gemm_bf16<<<gg, 256, 0, stream>>>(Xk, Wkb, k_b, Kp, nullptr, nullptr);
    gemm_bf16<<<gg, 256, 0, stream>>>(Xv, Wvb, v_b, nullptr, Vt, nullptr);
    attn_kernel<<<Bq * Hq * (Sq / 16), 128, 0, stream>>>(Qp, Kp, Vt, attn, Ctx);
    gemm_bf16<<<gg, 256, 0, stream>>>(Ctx, Wob, out_b, nullptr, nullptr, outp);
}
